// ROTPooling_17248588661359
// MI455X (gfx1250) — matrix operand order repaired and checked by reading, not compiled, not run on hardware
//
#include <hip/hip_runtime.h>
#include <hip/hip_bf16.h>
#include <math.h>

// ---------------------------------------------------------------------------
// ROT pooling (regularized OT Bregman-ADMM) for MI455X / gfx1250, wave32.
// All large GEMMs run on v_wmma_f32_16x16x32_bf16 (bf16 in, f32 accumulate),
// register-blocked 2x2 tiles per wave (32x32). K is a template constant
// (256 or 512) so the K-loop fully unrolls: the scheduler interleaves
// fragment loads with the WMMA stream with no loop-carried register copies.
// ---------------------------------------------------------------------------

#define BG        64          // NUM_GRAPHS
#define DIMD      256         // DIM
#define TOTN      16384       // TOTAL_NODES
#define NMAX      512         // N_MAX
#define NITERS    4
#define NEGV      (-1000000000.0f)

typedef __bf16 bf16;
typedef __attribute__((ext_vector_type(16))) __bf16 v16bf;
typedef __attribute__((ext_vector_type(8)))  __bf16 v8bf;
typedef __attribute__((ext_vector_type(8)))  float  v8f;

// ---------------- block reductions (256-thread blocks) ---------------------
__device__ __forceinline__ float blockMax(float v, float* sm) {
    int t = threadIdx.x;
    sm[t] = v; __syncthreads();
    for (int s = blockDim.x >> 1; s > 0; s >>= 1) {
        if (t < s) sm[t] = fmaxf(sm[t], sm[t + s]);
        __syncthreads();
    }
    float r = sm[0]; __syncthreads();
    return r;
}
__device__ __forceinline__ float blockSum(float v, float* sm) {
    int t = threadIdx.x;
    sm[t] = v; __syncthreads();
    for (int s = blockDim.x >> 1; s > 0; s >>= 1) {
        if (t < s) sm[t] += sm[t + s];
        __syncthreads();
    }
    float r = sm[0]; __syncthreads();
    return r;
}

// ---------------- WMMA fragment helpers ------------------------------------
// A 16x32 bf16 fragment, lane(l,h): elems 0..7 = A[l][h*8..+7],
//                                   elems 8..15 = A[l][16+h*8..+7]
__device__ __forceinline__ v16bf load_fragA(const bf16* __restrict__ p, int h) {
    v8bf lo = *reinterpret_cast<const v8bf*>(p + h * 8);
    v8bf hi = *reinterpret_cast<const v8bf*>(p + 16 + h * 8);
    v16bf f;
#pragma unroll
    for (int i = 0; i < 8; ++i) { f[i] = lo[i]; f[8 + i] = hi[i]; }
    return f;
}
// B 32x16 bf16 fragment from pre-transposed Bt, lane(l,h):
//   elems 0..7 = Bt[l][h*8..+7], elems 8..15 = Bt[l][16+h*8..+7]
__device__ __forceinline__ v16bf load_fragB(const bf16* __restrict__ p, int h) {
    v8bf lo = *reinterpret_cast<const v8bf*>(p + h * 8);
    v8bf hi = *reinterpret_cast<const v8bf*>(p + 16 + h * 8);
    v16bf f;
#pragma unroll
    for (int i = 0; i < 8; ++i) { f[i] = lo[i]; f[8 + i] = hi[i]; }
    return f;
}
#define WMMA_BF16(a, bb, c) __builtin_amdgcn_wmma_f32_16x16x32_bf16( \
        false, (a), false, (bb), (short)0, (c), false, false)

// ---------------- generic batched bf16 WMMA GEMM:  C = alpha * A * Bt^T ----
// A  : row-major (M x K) bf16, batch stride sA, leading dim lda
// Bt : row-major (N x K) bf16 (B pre-transposed), stride sB, ldb
// C  : fp32 and/or bf16 outputs (either may be null), stride sC, ldc
// grid = (M/64, N/128, batch), block = 256 threads = 8 waves (2M x 4N),
// each wave owns a 32x32 output (2x2 WMMA tiles). K-loop fully unrolled.
template<int K>
__global__ void __launch_bounds__(256)
gemm_bf16_wmma(const bf16* __restrict__ A,  long sA, int lda,
               const bf16* __restrict__ Bt, long sB, int ldb,
               float* __restrict__ Cf, bf16* __restrict__ Cb, long sC, int ldc,
               float alphaC, const float* __restrict__ alphaP)
{
    const int lane = threadIdx.x & 31;
    const int wave = threadIdx.x >> 5;
    const int wm   = wave >> 2;              // 0..1
    const int wn   = wave & 3;               // 0..3
    const int h    = lane >> 4;
    const int l    = lane & 15;
    const int b    = blockIdx.z;
    const int m0   = blockIdx.x * 64 + wm * 32;
    const int n0   = blockIdx.y * 128 + wn * 32;

    const bf16* Ar0 = A  + (long)b * sA + (long)(m0 + l) * lda;
    const bf16* Ar1 = Ar0 + 16 * lda;
    const bf16* Br0 = Bt + (long)b * sB + (long)(n0 + l) * ldb;
    const bf16* Br1 = Br0 + 16 * ldb;

    v8f acc00 = {}, acc01 = {}, acc10 = {}, acc11 = {};
#pragma unroll
    for (int k0 = 0; k0 < K; k0 += 32) {
        v16bf a0 = load_fragA(Ar0 + k0, h);
        v16bf a1 = load_fragA(Ar1 + k0, h);
        v16bf b0 = load_fragB(Br0 + k0, h);
        v16bf b1 = load_fragB(Br1 + k0, h);
        acc00 = WMMA_BF16(a0, b0, acc00);
        acc01 = WMMA_BF16(a0, b1, acc01);
        acc10 = WMMA_BF16(a1, b0, acc10);
        acc11 = WMMA_BF16(a1, b1, acc11);
    }

    float alpha = alphaC * (alphaP ? *alphaP : 1.0f);
    long cbase = (long)b * sC;
    // C 16x16 f32 tile layout: vgpr v, lane(l,h) -> row v+8*h, col l
#pragma unroll
    for (int v = 0; v < 8; ++v) {
        int r0 = m0 + v + 8 * h, r1 = r0 + 16;
        int c0 = n0 + l,         c1 = c0 + 16;
        float v00 = acc00[v] * alpha, v01 = acc01[v] * alpha;
        float v10 = acc10[v] * alpha, v11 = acc11[v] * alpha;
        if (Cf) {
            Cf[cbase + (long)r0 * ldc + c0] = v00;
            Cf[cbase + (long)r0 * ldc + c1] = v01;
            Cf[cbase + (long)r1 * ldc + c0] = v10;
            Cf[cbase + (long)r1 * ldc + c1] = v11;
        }
        if (Cb) {
            Cb[cbase + (long)r0 * ldc + c0] = (bf16)v00;
            Cb[cbase + (long)r0 * ldc + c1] = (bf16)v01;
            Cb[cbase + (long)r1 * ldc + c0] = (bf16)v10;
            Cb[cbase + (long)r1 * ldc + c1] = (bf16)v11;
        }
    }
}

// ---------------- scalars: softplus(a0..a3) + derived reciprocals ----------
__device__ __forceinline__ float softplusf(float x) {
    return x > 0.f ? x + log1pf(expf(-x)) : log1pf(expf(x));
}
__global__ void k_params(const float* a0, const float* a1, const float* a2,
                         const float* a3, float* prm) {
    if (threadIdx.x == 0) {
        float s0 = softplusf(a0[0]), s1 = softplusf(a1[0]);
        float s2 = softplusf(a2[0]), s3 = softplusf(a3[0]);
        prm[0] = s0; prm[1] = s1; prm[2] = s2; prm[3] = s3;
        prm[4] = 1.f / s3;          // 1/a3
        prm[5] = 1.f / (s0 + s3);   // 1/(a0+a3)
        prm[6] = 1.f / (s1 + s3);   // 1/(a1+a3)
    }
}

// ---------------- segment counts / offsets (batch is sorted) ---------------
__global__ void k_prep(const int* __restrict__ batch, int* counts, int* offsets) {
    __shared__ int c[BG];
    int t = threadIdx.x;
    if (t < BG) c[t] = 0;
    __syncthreads();
    for (int i = t; i < TOTN; i += blockDim.x) atomicAdd(&c[batch[i]], 1);
    __syncthreads();
    if (t == 0) {
        int off = 0;
        for (int g = 0; g < BG; ++g) { counts[g] = c[g]; offsets[g] = off; off += c[g]; }
    }
}

// ---------------- fp32 -> bf16 cast ---------------------------------------
__global__ void k_cast(const float* __restrict__ s, bf16* __restrict__ d, int n) {
    int i = blockIdx.x * blockDim.x + threadIdx.x;
    if (i < n) d[i] = (bf16)s[i];
}

// ---------------- scores = tanh(hidden) . w2 (one block / node) -----------
__global__ void __launch_bounds__(256)
k_score(const float* __restrict__ hid, const float* __restrict__ w2,
        float* __restrict__ scores) {
    __shared__ float sm[256];
    int nid = blockIdx.x, t = threadIdx.x;
    const float* h = hid + (long)nid * 512;
    float p = tanhf(h[t]) * w2[t] + tanhf(h[t + 256]) * w2[t + 256];
    float s = blockSum(p, sm);
    if (t == 0) scores[nid] = s;
}

// ---------------- per-graph segment softmax -> dense q0, log(q0+eps) ------
__global__ void __launch_bounds__(256)
k_q0(const float* __restrict__ scores, const int* counts, const int* offsets,
     float* __restrict__ q0d, float* __restrict__ lq0d) {
    __shared__ float sm[256];
    int b = blockIdx.x, t = threadIdx.x;
    int cnt = counts[b], off = offsets[b];
    float m = -3.4e38f;
    for (int p = t; p < cnt; p += 256) m = fmaxf(m, scores[off + p]);
    m = blockMax(m, sm);
    float s = 0.f;
    for (int p = t; p < cnt; p += 256) s += expf(scores[off + p] - m);
    s = blockSum(s, sm);
    float inv = 1.f / (s + 1e-16f);
    for (int p = t; p < NMAX; p += 256) {
        float q = (p < cnt) ? expf(scores[off + p] - m) * inv : 0.f;
        q0d[(long)b * NMAX + p]  = q;
        lq0d[(long)b * NMAX + p] = logf(q + 1e-8f);
    }
}

// ---------------- dense bf16 xd and xd^T (zero padded) --------------------
__global__ void k_dense(const float* __restrict__ x, const int* counts,
                        const int* offsets, bf16* __restrict__ xd,
                        bf16* __restrict__ xdT) {
    long i = (long)blockIdx.x * blockDim.x + threadIdx.x;
    if (i >= (long)BG * NMAX * DIMD) return;
    int d = (int)(i % DIMD); long r = i / DIMD;
    int n = (int)(r % NMAX); int b = (int)(r / NMAX);
    float v = (n < counts[b]) ? x[(long)(offsets[b] + n) * DIMD + d] : 0.f;
    xd[i] = (bf16)v;
    xdT[((long)b * DIMD + d) * NMAX + n] = (bf16)v;
}

// ---------------- ADMM state init -----------------------------------------
__global__ void k_init(const float* __restrict__ lq0d, float* __restrict__ log_t,
                       float* __restrict__ log_s, float* __restrict__ z,
                       bf16* __restrict__ sT, float* __restrict__ log_mu,
                       float* __restrict__ z1, float* __restrict__ log_eta,
                       float* __restrict__ z2) {
    long i = (long)blockIdx.x * blockDim.x + threadIdx.x;
    if (i >= (long)BG * NMAX * DIMD) return;
    const float LOGP0 = logf(1.0f / DIMD + 1e-8f);
    int d = (int)(i % DIMD); long r = i / DIMD;
    int n = (int)(r % NMAX); int b = (int)(r / NMAX);
    float lq  = lq0d[(long)b * NMAX + n];
    float lt0 = lq + LOGP0;
    log_t[i] = lt0; log_s[i] = lt0; z[i] = 0.f;
    sT[((long)b * DIMD + d) * NMAX + n] = (bf16)expf(lt0);
    if (d == 0) { log_mu[(long)b * NMAX + n] = lq;    z1[(long)b * NMAX + n] = 0.f; }
    if (n == 0) { log_eta[(long)b * DIMD + d] = LOGP0; z2[(long)b * DIMD + d] = 0.f; }
}

// ---------------- log_t = log_mu + log_softmax_axis2(y) -------------------
// y = (x - gw - z)/a3 + log_s ; one 256-thread block per (b,n) row.
__global__ void __launch_bounds__(256)
k_rowsoftmax(const float* __restrict__ x, const int* counts, const int* offsets,
             const float* __restrict__ gw, const float* __restrict__ z,
             const float* __restrict__ log_s, const float* __restrict__ log_mu,
             float* __restrict__ log_t, const float* __restrict__ prm) {
    __shared__ float sm[256];
    int n = blockIdx.x, b = blockIdx.y, d = threadIdx.x;
    long base = ((long)b * NMAX + n) * DIMD;
    int cnt = counts[b];
    float xv = (n < cnt) ? x[(long)(offsets[b] + n) * DIMD + d] : 0.f;
    float y  = (xv - gw[base + d] - z[base + d]) * prm[4] + log_s[base + d];
    float m  = blockMax(y, sm);
    float S  = blockSum(expf(y - m), sm);
    float lse = m + logf(S);
    // logsumexp(log_t, axis2) == log_mu exactly (log-softmax sums to 1)
    log_t[base + d] = log_mu[(long)b * NMAX + n] + (y - lse);
}

// ---------------- log_s = log_eta + log_softmax_axis1(masked y2) ----------
// y2 = (a3*log_t - z)/(a0+a3), NEG on padded rows.
// 1024 threads/block: 256 columns x 4-way split of the 512-row loop,
// partial online logsumexps merged through LDS.
__global__ void __launch_bounds__(1024)
k_colsoftmax(const float* __restrict__ log_t, const float* __restrict__ z,
             const float* __restrict__ log_eta, const int* counts,
             float* __restrict__ log_s, const float* __restrict__ prm) {
    __shared__ float smM[1024], smS[1024], sLse[256];
    int tid = threadIdx.x;
    int d = tid & 255, q = tid >> 8;
    int b = blockIdx.x;
    int cnt = counts[b];
    float a3 = prm[3], inv = prm[5];
    long base = (long)b * NMAX * DIMD + d;
    float m = NEGV, S = 0.f;
    for (int n = q * 128; n < q * 128 + 128; ++n) {     // online logsumexp
        long i = base + (long)n * DIMD;
        float y = (n < cnt) ? (a3 * log_t[i] - z[i]) * inv : NEGV;
        if (y > m) { S = S * expf(m - y) + 1.f; m = y; }
        else         S += expf(y - m);
    }
    smM[tid] = m; smS[tid] = S; __syncthreads();
    if (q == 0) {
#pragma unroll
        for (int j = 1; j < 4; ++j) {
            float m2 = smM[d + j * 256], S2 = smS[d + j * 256];
            if (m2 > m) { S = S * expf(m - m2) + S2; m = m2; }
            else          S += S2 * expf(m2 - m);
        }
        sLse[d] = m + logf(S);
    }
    __syncthreads();
    float lse = sLse[d];
    float le  = log_eta[(long)b * DIMD + d];
    for (int n = q * 128; n < q * 128 + 128; ++n) {
        long i = base + (long)n * DIMD;
        float y = (n < cnt) ? (a3 * log_t[i] - z[i]) * inv : NEGV;
        log_s[i] = le + (y - lse);
    }
}

// ---------------- marginal duals (uses log_tsum==log_mu, log_ssum==log_eta)
__global__ void __launch_bounds__(512)
k_marginals(const float* __restrict__ lq0d, float* __restrict__ log_mu,
            float* __restrict__ z1, float* __restrict__ log_eta,
            float* __restrict__ z2, const float* __restrict__ prm) {
    int b = blockIdx.x, t = threadIdx.x;
    float a1 = prm[1], a3 = prm[3], inv = prm[6];
    const float LOGP0 = logf(1.0f / DIMD + 1e-8f);
    if (t < NMAX) {
        long i = (long)b * NMAX + t;
        float old = log_mu[i];                         // == log_tsum
        float nw  = (a1 * lq0d[i] + a3 * old - z1[i]) * inv;
        z1[i] += a3 * (expf(nw) - expf(old));
        log_mu[i] = nw;
    }
    if (t < DIMD) {
        long i = (long)b * DIMD + t;
        float old = log_eta[i];                        // == log_ssum
        float nw  = (a1 * LOGP0 + a3 * old - z2[i]) * inv;
        z2[i] += a3 * (expf(nw) - expf(old));
        log_eta[i] = nw;
    }
}

// ---------------- z += a3*(exp(log_t)-exp(log_s)); sT_bf = exp(log_s)^T ---
__global__ void k_dual(const float* __restrict__ log_t, const float* __restrict__ log_s,
                       float* __restrict__ z, bf16* __restrict__ sT,
                       const float* __restrict__ prm) {
    long i = (long)blockIdx.x * blockDim.x + threadIdx.x;
    if (i >= (long)BG * NMAX * DIMD) return;
    float t = expf(log_t[i]), s = expf(log_s[i]);
    z[i] += prm[3] * (t - s);
    int d = (int)(i % DIMD); long r = i / DIMD;
    int n = (int)(r % NMAX); int b = (int)(r / NMAX);
    sT[((long)b * DIMD + d) * NMAX + n] = (bf16)s;
}

// ---------------- out[b][d] = D * sum_n xd * exp(log_t) * mask ------------
// 1024 threads/block: 256 columns x 4-way row split, LDS combine.
__global__ void __launch_bounds__(1024)
k_out(const float* __restrict__ x, const float* __restrict__ log_t,
      const int* counts, const int* offsets, float* __restrict__ out) {
    __shared__ float sm[1024];
    int tid = threadIdx.x;
    int d = tid & 255, q = tid >> 8;
    int b = blockIdx.x;
    int cnt = counts[b], off = offsets[b];
    float acc = 0.f;
    for (int p = q; p < cnt; p += 4)
        acc += x[(long)(off + p) * DIMD + d] *
               expf(log_t[((long)b * NMAX + p) * DIMD + d]);
    sm[tid] = acc; __syncthreads();
    if (q == 0)
        out[(long)b * DIMD + d] =
            (float)DIMD * (sm[d] + sm[d + 256] + sm[d + 512] + sm[d + 768]);
}

// ===========================================================================
extern "C" void kernel_launch(void* const* d_in, const int* in_sizes, int n_in,
                              void* d_out, int out_size, void* d_ws, size_t ws_size,
                              hipStream_t stream) {
    const float* x  = (const float*)d_in[0];
    const float* w1 = (const float*)d_in[1];
    const float* w2 = (const float*)d_in[2];
    const float* a0 = (const float*)d_in[3];
    const float* a1 = (const float*)d_in[4];
    const float* a2 = (const float*)d_in[5];
    const float* a3 = (const float*)d_in[6];
    const int* batch = (const int*)d_in[7];
    float* out = (float*)d_out;

    // ---- workspace bump allocator (256B aligned for 128b vector loads) ----
    char* wp = (char*)d_ws;
    auto alloc = [&](size_t bytes) -> void* {
        void* r = (void*)wp; wp += (bytes + 255) & ~(size_t)255; return r;
    };
    const long BND = (long)BG * NMAX * DIMD;       // 8.4M elems
    int*   counts  = (int*)alloc(BG * 4);
    int*   offsets = (int*)alloc(BG * 4);
    float* prm     = (float*)alloc(8 * 4);
    float* scores  = (float*)alloc(TOTN * 4);
    float* q0d     = (float*)alloc((long)BG * NMAX * 4);
    float* lq0d    = (float*)alloc((long)BG * NMAX * 4);
    bf16*  x_bf    = (bf16*)alloc((long)TOTN * DIMD * 2);
    bf16*  w1_bf   = (bf16*)alloc((long)512 * DIMD * 2);
    bf16*  xd_bf   = (bf16*)alloc(BND * 2);
    bf16*  xdT_bf  = (bf16*)alloc(BND * 2);
    bf16*  c1_bf   = (bf16*)alloc((long)BG * DIMD * DIMD * 2);
    bf16*  c2_bf   = (bf16*)alloc((long)BG * NMAX * NMAX * 2);
    bf16*  sT_bf   = (bf16*)alloc(BND * 2);
    bf16*  tmp_bf  = (bf16*)alloc(BND * 2);
    float* fbig    = (float*)alloc(BND * 4);       // hidden (16384x512) then gw
    float* log_t   = (float*)alloc(BND * 4);
    float* log_s   = (float*)alloc(BND * 4);
    float* zbuf    = (float*)alloc(BND * 4);
    float* log_mu  = (float*)alloc((long)BG * NMAX * 4);
    float* z1      = (float*)alloc((long)BG * NMAX * 4);
    float* log_eta = (float*)alloc((long)BG * DIMD * 4);
    float* z2      = (float*)alloc((long)BG * DIMD * 4);

    const int eb = 256;
    const long ebn = (BND + eb - 1) / eb;

    // ---- prologue ----
    k_params<<<1, 32, 0, stream>>>(a0, a1, a2, a3, prm);
    k_prep<<<1, 256, 0, stream>>>(batch, counts, offsets);
    k_cast<<<(TOTN * DIMD + eb - 1) / eb, eb, 0, stream>>>(x, x_bf, TOTN * DIMD);
    k_cast<<<(512 * DIMD + eb - 1) / eb, eb, 0, stream>>>(w1, w1_bf, 512 * DIMD);

    // hidden = x @ w1^T   (16384x512, K=256); w1 is already the Bt layout
    gemm_bf16_wmma<256><<<dim3(TOTN / 64, 512 / 128, 1), 256, 0, stream>>>(
        x_bf, 0, DIMD, w1_bf, 0, DIMD, fbig, nullptr, 0, 512, 1.f, nullptr);
    k_score<<<TOTN, 256, 0, stream>>>(fbig, w2, scores);
    k_q0<<<BG, 256, 0, stream>>>(scores, counts, offsets, q0d, lq0d);
    k_dense<<<ebn, eb, 0, stream>>>(x, counts, offsets, xd_bf, xdT_bf);

    // c2 = xd xd^T / D  (512x512, K=256, symmetric -> Bt = xd)
    gemm_bf16_wmma<256><<<dim3(NMAX / 64, NMAX / 128, BG), 256, 0, stream>>>(
        xd_bf, (long)NMAX * DIMD, DIMD, xd_bf, (long)NMAX * DIMD, DIMD,
        nullptr, c2_bf, (long)NMAX * NMAX, NMAX, 1.f / DIMD, nullptr);
    // c1 = xd^T xd / N  (256x256, K=512, symmetric -> operands are xdT)
    gemm_bf16_wmma<512><<<dim3(DIMD / 64, DIMD / 128, BG), 256, 0, stream>>>(
        xdT_bf, (long)DIMD * NMAX, NMAX, xdT_bf, (long)DIMD * NMAX, NMAX,
        nullptr, c1_bf, (long)DIMD * DIMD, DIMD, 1.f / NMAX, nullptr);

    k_init<<<ebn, eb, 0, stream>>>(lq0d, log_t, log_s, zbuf, sT_bf,
                                   log_mu, z1, log_eta, z2);

    // ---- Bregman-ADMM iterations ----
    for (int it = 0; it < NITERS; ++it) {
        // tmp = c2 @ s   (512x256, K=512; Bt = s^T, symmetric c2 as A)
        gemm_bf16_wmma<512><<<dim3(NMAX / 64, DIMD / 128, BG), 256, 0, stream>>>(
            c2_bf, (long)NMAX * NMAX, NMAX, sT_bf, (long)DIMD * NMAX, NMAX,
            nullptr, tmp_bf, (long)NMAX * DIMD, DIMD, 1.f, nullptr);
        // gw = a2 * tmp @ c1  (512x256, K=256; c1 symmetric -> Bt = c1)
        gemm_bf16_wmma<256><<<dim3(NMAX / 64, DIMD / 128, BG), 256, 0, stream>>>(
            tmp_bf, (long)NMAX * DIMD, DIMD, c1_bf, (long)DIMD * DIMD, DIMD,
            fbig, nullptr, (long)NMAX * DIMD, DIMD, 1.f, prm + 2);

        k_rowsoftmax<<<dim3(NMAX, BG), 256, 0, stream>>>(
            x, counts, offsets, fbig, zbuf, log_s, log_mu, log_t, prm);
        k_colsoftmax<<<BG, 1024, 0, stream>>>(log_t, zbuf, log_eta, counts, log_s, prm);
        k_marginals<<<BG, 512, 0, stream>>>(lq0d, log_mu, z1, log_eta, z2, prm);
        k_dual<<<ebn, eb, 0, stream>>>(log_t, log_s, zbuf, sT_bf, prm);
    }

    // ---- epilogue ----
    k_out<<<BG, 1024, 0, stream>>>(x, log_t, counts, offsets, out);
}
